// SimpleMoE_6751688589360
// MI455X (gfx1250) — matrix operand order repaired and checked by reading, not compiled, not run on hardware
//
#include <hip/hip_runtime.h>
#include <hip/hip_bf16.h>
#include <math.h>

#define HIDDEN 1024
#define FF     4096
#define NEXP   8
#define TOKENS 8192   // B*S = 4*2048

#define BM 128
#define BN 128
#define KT 32
#define LDSS 40       // LDS row stride in bf16 elements (80B) -> conflict-friendly, 16B aligned

typedef __bf16 bf16_t;
typedef __attribute__((ext_vector_type(16))) __bf16 bf16x16;
typedef __attribute__((ext_vector_type(8)))  __bf16 bf16x8;
typedef __attribute__((ext_vector_type(4)))  __bf16 bf16x4;
typedef __attribute__((ext_vector_type(8)))  float  floatx8;
typedef int v4i_t __attribute__((vector_size(16)));

// ---- gfx1250 async global->LDS path (ASYNCcnt), with compile-time fallback ----
#if defined(__has_builtin)
#if __has_builtin(__builtin_amdgcn_global_load_async_to_lds_b128) && \
    __has_builtin(__builtin_amdgcn_s_wait_asynccnt)
#define HAVE_ASYNC_LDS 1
#endif
#endif

__device__ inline void async_copy16(void* lds_dst, const void* gsrc) {
#ifdef HAVE_ASYNC_LDS
    __builtin_amdgcn_global_load_async_to_lds_b128(
        (__attribute__((address_space(1))) v4i_t*)gsrc,
        (__attribute__((address_space(3))) v4i_t*)lds_dst, 0, 0);
#else
    *reinterpret_cast<bf16x8*>(lds_dst) = *reinterpret_cast<const bf16x8*>(gsrc);
#endif
}

__device__ inline void async_wait_all() {
#ifdef HAVE_ASYNC_LDS
    __builtin_amdgcn_s_wait_asynccnt(0);
#endif
}

__device__ inline floatx8 fzero8() {
    floatx8 z;
#pragma unroll
    for (int i = 0; i < 8; ++i) z[i] = 0.0f;
    return z;
}

// A-matrix fragment (16x32 bf16, ISA 7.12.2): lane r: K {0..7, 16..23}; lane r+16: K {8..15, 24..31}
__device__ inline bf16x16 frag_a(const bf16_t* lds, int tileRow, int lane) {
    const int hs = (lane >> 4) & 1;
    const int r  = lane & 15;
    const bf16_t* p = lds + (tileRow + r) * LDSS;
    bf16x8 lo = *reinterpret_cast<const bf16x8*>(p + hs * 8);
    bf16x8 hi = *reinterpret_cast<const bf16x8*>(p + 16 + hs * 8);
    bf16x16 f;
#pragma unroll
    for (int i = 0; i < 8; ++i) { f[i] = lo[i]; f[8 + i] = hi[i]; }
    return f;
}

// B-matrix fragment (32x16 bf16): lane n: col n, K {0..7, 16..23}; lane n+16: col n, K {8..15, 24..31}.
// Tile stored as [N][K] rows (w is row-major [N][K]), so each lane reads two 16B chunks 32B apart.
__device__ inline bf16x16 frag_b(const bf16_t* lds, int tileRow, int lane) {
    const int hs = (lane >> 4) & 1;
    const int n  = lane & 15;
    const bf16_t* p = lds + (tileRow + n) * LDSS + hs * 8;
    bf16x8 lo = *reinterpret_cast<const bf16x8*>(p);
    bf16x8 hi = *reinterpret_cast<const bf16x8*>(p + 16);
    bf16x16 f;
#pragma unroll
    for (int i = 0; i < 8; ++i) { f[i] = lo[i]; f[8 + i] = hi[i]; }
    return f;
}

// ---------------- init ----------------
__global__ void moe_init(float* __restrict__ out, size_t n,
                         int* __restrict__ counts, float* __restrict__ probs) {
    size_t i = (size_t)blockIdx.x * blockDim.x + threadIdx.x;
    if (i < n) out[i] = 0.0f;
    if (i < NEXP) { counts[i] = 0; probs[i] = 0.0f; }
}

// ---------------- f32 -> bf16 ----------------
__global__ void convert_f32_bf16(const float* __restrict__ src, bf16_t* __restrict__ dst, size_t n) {
    size_t i = ((size_t)blockIdx.x * blockDim.x + threadIdx.x) * 4;
    if (i < n) {
        float4 v = *reinterpret_cast<const float4*>(src + i);
        bf16x4 b;
        b[0] = (bf16_t)v.x; b[1] = (bf16_t)v.y; b[2] = (bf16_t)v.z; b[3] = (bf16_t)v.w;
        *reinterpret_cast<bf16x4*>(dst + i) = b;
    }
}

// ---------------- gating: logits, top-2, routing lists, softmax probs ----------------
__global__ __launch_bounds__(256)
void moe_gate(const float* __restrict__ x, const float* __restrict__ gw,
              const float* __restrict__ gb, int* __restrict__ counts,
              int* __restrict__ tok_list, float* __restrict__ tok_w,
              float* __restrict__ probs_accum) {
    __shared__ float sgw[NEXP * HIDDEN];   // 32KB
    __shared__ float sp[NEXP];
    const int tid = threadIdx.x;
    for (int i = tid; i < NEXP * HIDDEN; i += 256) sgw[i] = gw[i];
    if (tid < NEXP) sp[tid] = 0.0f;
    __syncthreads();

    const int tok = blockIdx.x * 256 + tid;
    float a[NEXP];
#pragma unroll
    for (int e = 0; e < NEXP; ++e) a[e] = gb[e];
    const float* xr = x + (size_t)tok * HIDDEN;
    for (int k = 0; k < HIDDEN; k += 4) {
        float4 xv = *reinterpret_cast<const float4*>(xr + k);
#pragma unroll
        for (int e = 0; e < NEXP; ++e) {
            const float* g = &sgw[e * HIDDEN + k];
            a[e] += xv.x * g[0] + xv.y * g[1] + xv.z * g[2] + xv.w * g[3];
        }
    }
    // top-2 (first-occurrence tie-break like lax.top_k)
    int e0 = 0; float v0 = a[0];
#pragma unroll
    for (int e = 1; e < NEXP; ++e) if (a[e] > v0) { v0 = a[e]; e0 = e; }
    int e1 = 0; float v1 = -3.4e38f;
#pragma unroll
    for (int e = 0; e < NEXP; ++e) if (e != e0 && a[e] > v1) { v1 = a[e]; e1 = e; }
    float p1 = __expf(v1 - v0);
    float inv = 1.0f / (1.0f + p1);
    float w0 = inv, w1 = p1 * inv;
    // full softmax for l_aux probs
    float s = 0.0f, pe[NEXP];
#pragma unroll
    for (int e = 0; e < NEXP; ++e) { pe[e] = __expf(a[e] - v0); s += pe[e]; }
    float invs = 1.0f / s;
#pragma unroll
    for (int e = 0; e < NEXP; ++e) atomicAdd(&sp[e], pe[e] * invs);
    int pos0 = atomicAdd(&counts[e0], 1);
    tok_list[e0 * TOKENS + pos0] = tok; tok_w[e0 * TOKENS + pos0] = w0;
    int pos1 = atomicAdd(&counts[e1], 1);
    tok_list[e1 * TOKENS + pos1] = tok; tok_w[e1 * TOKENS + pos1] = w1;
    __syncthreads();
    if (tid < NEXP) atomicAdd(&probs_accum[tid], sp[tid]);
}

// ---------------- offsets scan + l_aux ----------------
__global__ void moe_finalize(const int* __restrict__ counts, int* __restrict__ offsets,
                             const float* __restrict__ probs_accum, float* __restrict__ laux) {
    if (threadIdx.x == 0 && blockIdx.x == 0) {
        int off = 0;
        for (int e = 0; e < NEXP; ++e) { offsets[e] = off; off += counts[e]; }
        float s = 0.0f;
        for (int e = 0; e < NEXP; ++e) {
            float p = probs_accum[e] / (float)TOKENS;
            s += p * p;
        }
        *laux = s * (float)NEXP;
    }
}

// ---------------- GEMM1: gathered x_bf16 @ w1[e]^T -> GELU -> h (bf16 arena) ----------------
__global__ __launch_bounds__(256)
void moe_gemm1(const bf16_t* __restrict__ xb, const bf16_t* __restrict__ w1b,
               const float* __restrict__ b1, bf16_t* __restrict__ h,
               const int* __restrict__ counts, const int* __restrict__ offsets,
               const int* __restrict__ tok_list) {
    const int e     = blockIdx.z;
    const int cnt   = counts[e];
    const int mBase = blockIdx.y * BM;
    if (mBase >= cnt) return;
    const int nBase = blockIdx.x * BN;
    const int off   = offsets[e];
    const int eT    = e * TOKENS;

    __shared__ bf16_t ldsA[2 * BM * LDSS];
    __shared__ bf16_t ldsB[2 * BN * LDSS];

    const int tid  = threadIdx.x;
    const int lane = tid & 31;
    const int wid  = tid >> 5;
    const int wm   = (wid & 3) * 32;   // wave M offset
    const int wn   = (wid >> 2) * 64;  // wave N offset

    floatx8 acc[2][4];
#pragma unroll
    for (int mi = 0; mi < 2; ++mi)
#pragma unroll
        for (int ni = 0; ni < 4; ++ni) acc[mi][ni] = fzero8();

    // ---- hoisted, branch-free gather: per-thread global row pointers (K-invariant) ----
    const int tr = tid >> 2;          // 0..63  (row within pass)
    const int tc = (tid & 3) * 8;     // bf16 col (16B chunk)
    const bf16_t* aga[2];
    const bf16_t* bga[2];
    int ldst[2];
    const bf16_t* wptr = w1b + (size_t)e * FF * HIDDEN;
#pragma unroll
    for (int p = 0; p < 2; ++p) {
        int gr  = mBase + p * 64 + tr;
        int grc = (gr < cnt) ? gr : (cnt - 1);           // clamp: dup rows are discarded in epilogue
        aga[p]  = xb + (size_t)tok_list[eT + grc] * HIDDEN + tc;
        bga[p]  = wptr + (size_t)(nBase + p * 64 + tr) * HIDDEN + tc;
        ldst[p] = (p * 64 + tr) * LDSS + tc;
    }

    auto load_stage = [&](int stage, int kt) {
        const int so = stage * BM * LDSS;
#pragma unroll
        for (int p = 0; p < 2; ++p) {
            async_copy16(&ldsA[so + ldst[p]], aga[p] + kt);
            async_copy16(&ldsB[so + ldst[p]], bga[p] + kt);
        }
    };

    const int NK = HIDDEN / KT;
    load_stage(0, 0);
    async_wait_all();
    __syncthreads();

    for (int k = 0; k < NK; ++k) {
        const int cur = k & 1;
        if (k + 1 < NK) load_stage(cur ^ 1, (k + 1) * KT);   // overlap with WMMAs below
        const bf16_t* aB = ldsA + cur * BM * LDSS;
        const bf16_t* bB = ldsB + cur * BN * LDSS;
        bf16x16 af[2], bfr[4];
#pragma unroll
        for (int mi = 0; mi < 2; ++mi) af[mi] = frag_a(aB, wm + mi * 16, lane);
#pragma unroll
        for (int ni = 0; ni < 4; ++ni) bfr[ni] = frag_b(bB, wn + ni * 16, lane);
#pragma unroll
        for (int mi = 0; mi < 2; ++mi)
#pragma unroll
            for (int ni = 0; ni < 4; ++ni)
                acc[mi][ni] = __builtin_amdgcn_wmma_f32_16x16x32_bf16(
                    false, af[mi], false, bfr[ni], (short)0, acc[mi][ni], false, false);
        async_wait_all();
        __syncthreads();
    }

    // epilogue: + b1, exact GELU, store bf16
    const int hs = lane >> 4, ln = lane & 15;
#pragma unroll
    for (int mi = 0; mi < 2; ++mi)
#pragma unroll
        for (int ni = 0; ni < 4; ++ni) {
#pragma unroll
            for (int j = 0; j < 8; ++j) {
                int ml = wm + mi * 16 + hs * 8 + j;
                int gm = mBase + ml;
                if (gm < cnt) {
                    int gn = nBase + wn + ni * 16 + ln;
                    float v = acc[mi][ni][j] + b1[e * FF + gn];
                    v = 0.5f * v * (1.0f + erff(v * 0.70710678118f));
                    h[(size_t)(off + gm) * FF + gn] = (bf16_t)v;
                }
            }
        }
}

// ---------------- GEMM2: h @ w2[e]^T -> +b2 -> *routing weight -> scatter-add out ----------------
__global__ __launch_bounds__(256)
void moe_gemm2(const bf16_t* __restrict__ h, const bf16_t* __restrict__ w2b,
               const float* __restrict__ b2, float* __restrict__ out,
               const int* __restrict__ counts, const int* __restrict__ offsets,
               const int* __restrict__ tok_list, const float* __restrict__ tok_w) {
    const int e     = blockIdx.z;
    const int cnt   = counts[e];
    const int mBase = blockIdx.y * BM;
    if (mBase >= cnt) return;
    const int nBase = blockIdx.x * BN;
    const int off   = offsets[e];
    const int eT    = e * TOKENS;

    __shared__ bf16_t ldsA[2 * BM * LDSS];
    __shared__ bf16_t ldsB[2 * BN * LDSS];

    const int tid  = threadIdx.x;
    const int lane = tid & 31;
    const int wid  = tid >> 5;
    const int wm   = (wid & 3) * 32;
    const int wn   = (wid >> 2) * 64;

    floatx8 acc[2][4];
#pragma unroll
    for (int mi = 0; mi < 2; ++mi)
#pragma unroll
        for (int ni = 0; ni < 4; ++ni) acc[mi][ni] = fzero8();

    const int tr = tid >> 2;
    const int tc = (tid & 3) * 8;
    const bf16_t* aga[2];
    const bf16_t* bga[2];
    int ldst[2];
    const bf16_t* wptr = w2b + (size_t)e * HIDDEN * FF;
#pragma unroll
    for (int p = 0; p < 2; ++p) {
        int gr  = mBase + p * 64 + tr;
        int grc = (gr < cnt) ? gr : (cnt - 1);
        aga[p]  = h + (size_t)(off + grc) * FF + tc;
        bga[p]  = wptr + (size_t)(nBase + p * 64 + tr) * FF + tc;
        ldst[p] = (p * 64 + tr) * LDSS + tc;
    }

    auto load_stage = [&](int stage, int kt) {
        const int so = stage * BM * LDSS;
#pragma unroll
        for (int p = 0; p < 2; ++p) {
            async_copy16(&ldsA[so + ldst[p]], aga[p] + kt);
            async_copy16(&ldsB[so + ldst[p]], bga[p] + kt);
        }
    };

    const int NK = FF / KT;
    load_stage(0, 0);
    async_wait_all();
    __syncthreads();

    for (int k = 0; k < NK; ++k) {
        const int cur = k & 1;
        if (k + 1 < NK) load_stage(cur ^ 1, (k + 1) * KT);
        const bf16_t* aB = ldsA + cur * BM * LDSS;
        const bf16_t* bB = ldsB + cur * BN * LDSS;
        bf16x16 af[2], bfr[4];
#pragma unroll
        for (int mi = 0; mi < 2; ++mi) af[mi] = frag_a(aB, wm + mi * 16, lane);
#pragma unroll
        for (int ni = 0; ni < 4; ++ni) bfr[ni] = frag_b(bB, wn + ni * 16, lane);
#pragma unroll
        for (int mi = 0; mi < 2; ++mi)
#pragma unroll
            for (int ni = 0; ni < 4; ++ni)
                acc[mi][ni] = __builtin_amdgcn_wmma_f32_16x16x32_bf16(
                    false, af[mi], false, bfr[ni], (short)0, acc[mi][ni], false, false);
        async_wait_all();
        __syncthreads();
    }

    const int hs = lane >> 4, ln = lane & 15;
#pragma unroll
    for (int mi = 0; mi < 2; ++mi)
#pragma unroll
        for (int ni = 0; ni < 4; ++ni) {
#pragma unroll
            for (int j = 0; j < 8; ++j) {
                int ml = wm + mi * 16 + hs * 8 + j;
                int gm = mBase + ml;
                if (gm < cnt) {
                    int gn = nBase + wn + ni * 16 + ln;
                    float v = acc[mi][ni][j] + b2[e * HIDDEN + gn];
                    int   tk = tok_list[eT + gm];
                    float rw = tok_w[eT + gm];
                    atomicAdd(&out[(size_t)tk * HIDDEN + gn], v * rw);
                }
            }
        }
}

extern "C" void kernel_launch(void* const* d_in, const int* in_sizes, int n_in,
                              void* d_out, int out_size, void* d_ws, size_t ws_size,
                              hipStream_t stream) {
    const float* x  = (const float*)d_in[0];
    const float* gw = (const float*)d_in[1];
    const float* gb = (const float*)d_in[2];
    const float* w1 = (const float*)d_in[3];
    const float* b1 = (const float*)d_in[4];
    const float* w2 = (const float*)d_in[5];
    const float* b2 = (const float*)d_in[6];
    float* out = (float*)d_out;

    char* ws = (char*)d_ws;
    size_t o = 0;
    auto carve = [&](size_t bytes) -> void* {
        void* p = ws + o;
        o += (bytes + 255) & ~(size_t)255;
        return p;
    };
    bf16_t* w1b      = (bf16_t*)carve((size_t)NEXP * FF * HIDDEN * 2);
    bf16_t* w2b      = (bf16_t*)carve((size_t)NEXP * HIDDEN * FF * 2);
    bf16_t* xb       = (bf16_t*)carve((size_t)TOKENS * HIDDEN * 2);
    bf16_t* hbuf     = (bf16_t*)carve((size_t)2 * TOKENS * FF * 2);   // arena: exactly 2T rows
    int*    tok_list = (int*)carve((size_t)NEXP * TOKENS * 4);
    float*  tok_w    = (float*)carve((size_t)NEXP * TOKENS * 4);
    int*    counts   = (int*)carve(64);
    int*    offsets  = (int*)carve(64);
    float*  probs    = (float*)carve(64);

    const size_t outN = (size_t)TOKENS * HIDDEN;
    moe_init<<<(unsigned)((outN + 255) / 256), 256, 0, stream>>>(out, outN, counts, probs);

    const size_t wn = (size_t)NEXP * FF * HIDDEN;
    convert_f32_bf16<<<(unsigned)((wn / 4 + 255) / 256), 256, 0, stream>>>(w1, w1b, wn);
    convert_f32_bf16<<<(unsigned)((wn / 4 + 255) / 256), 256, 0, stream>>>(w2, w2b, wn);
    convert_f32_bf16<<<(unsigned)((outN / 4 + 255) / 256), 256, 0, stream>>>(x, xb, outN);

    moe_gate<<<TOKENS / 256, 256, 0, stream>>>(x, gw, gb, counts, tok_list, tok_w, probs);
    moe_finalize<<<1, 32, 0, stream>>>(counts, offsets, probs, out + outN);

    dim3 g1(FF / BN, TOKENS / BM, NEXP);
    moe_gemm1<<<g1, 256, 0, stream>>>(xb, w1b, b1, hbuf, counts, offsets, tok_list);

    dim3 g2(HIDDEN / BN, TOKENS / BM, NEXP);
    moe_gemm2<<<g2, 256, 0, stream>>>(hbuf, w2b, b2, out, counts, offsets, tok_list, tok_w);
}
